// DGraphAttention_79096117723502
// MI455X (gfx1250) — compile-verified, matrix operand order checked by reading, not run on hardware
//
#include <hip/hip_runtime.h>
#include <hip/hip_bf16.h>
#include <cmath>

typedef __attribute__((ext_vector_type(16))) __bf16 v16bf;
typedef __attribute__((ext_vector_type(8)))  float  v8f;
typedef unsigned short u16b;

// ---------- helpers ----------
__device__ __forceinline__ u16b f2bf(float f) {
    union { float f; unsigned u; } x; x.f = f;
    unsigned r = x.u + 0x7FFFu + ((x.u >> 16) & 1u);   // round-nearest-even
    return (u16b)(r >> 16);
}

// Load one 16-bit WMMA operand fragment (A layout; B is loaded symmetrically
// from the K-major transposed operand). `p` points at row*ld + k0 + half*8.
// Two contiguous 16-byte chunks: K = {0..7} and K = {16..23} (+half*8).
__device__ __forceinline__ v16bf load_frag(const u16b* p) {
    union { uint4 q[2]; v16bf v; } f;
    f.q[0] = *(const uint4*)(p);
    f.q[1] = *(const uint4*)(p + 16);
    return f.v;
}

__device__ __forceinline__ v8f wmma_bf16(v16bf a, v16bf b, v8f c) {
    return __builtin_amdgcn_wmma_f32_16x16x32_bf16(
        /*neg_a=*/false, a, /*neg_b=*/false, b,
        /*c_mod=*/(short)0, c, /*reuse_a=*/false, /*reuse_b=*/false);
}

__device__ __forceinline__ void atomicMaxF(float* a, float v) {
    if (v >= 0.0f) atomicMax((int*)a, __float_as_int(v));
    else           atomicMin((unsigned int*)a, (unsigned int)__float_as_int(v));
}

// ---------- elementwise kernels ----------
__global__ void dga_cvt_bf16(const float* __restrict__ src, u16b* __restrict__ dst, int n) {
    for (int i = blockIdx.x * 256 + threadIdx.x; i < n; i += gridDim.x * 256)
        dst[i] = f2bf(src[i]);
}

__global__ void dga_init_stats(float* colmax, float* colsum, int E) {
    int i = blockIdx.x * 256 + threadIdx.x;
    if (i < E) { colmax[i] = -INFINITY; colsum[i] = 0.0f; }
}

// vgT[h][e] = bf16(v_full[src[e]][h])  (K-major layout for phase-2 B operand)
__global__ void dga_gather_vT(const float* __restrict__ vfull, const int* __restrict__ src,
                              u16b* __restrict__ vgT, int E, int H) {
    int idx = blockIdx.x * 256 + threadIdx.x;          // e fast-varying: coalesced writes
    if (idx >= E * H) return;
    int e = idx % E, h = idx / E;
    vgT[idx] = f2bf(vfull[(long)src[e] * H + h]);
}

// ---------- projection GEMM:  out[m, n] = sum_k A[row(m), k] * W[n, k] + bias[n] ----------
// A rows optionally gathered through rowIdx.  outF (f32) xor outB (bf16, pre-scaled).
__global__ void dga_proj(const u16b* __restrict__ X, const u16b* __restrict__ W,
                         const float* __restrict__ bias, const int* __restrict__ rowIdx,
                         int H, float scale,
                         float* __restrict__ outF, u16b* __restrict__ outB) {
    const int wave = threadIdx.x >> 5, lane = threadIdx.x & 31;
    const int half = lane >> 4, r16 = lane & 15;
    const int ntiles = H >> 4;
    const int tile = blockIdx.x * 8 + wave;            // grid sized exactly
    const int mt = tile / ntiles, nt = tile % ntiles;

    int arow = mt * 16 + r16;
    if (rowIdx) arow = rowIdx[arow];
    const u16b* aptr = X + (long)arow * H + half * 8;
    const u16b* bptr = W + (long)(nt * 16 + r16) * H + half * 8;

    v8f acc = {};
#pragma unroll 4
    for (int k = 0; k < H; k += 32)
        acc = wmma_bf16(load_frag(aptr + k), load_frag(bptr + k), acc);

    const int col = nt * 16 + r16;
    const float bb = bias[col];
    if (outF) {
#pragma unroll
        for (int r = 0; r < 8; ++r)
            outF[(long)(mt * 16 + r + half * 8) * H + col] = acc[r] + bb;
    } else {
#pragma unroll
        for (int r = 0; r < 8; ++r)
            outB[(long)(mt * 16 + r + half * 8) * H + col] = f2bf((acc[r] + bb) * scale);
    }
}

// ---------- column stats over logits[t,s] = qg[t]·kg[s]  (softmax axis = t) ----------
// mode 0: colmax[s] = max_t ; mode 1: colsum[s] = sum_t exp(l - colmax[s])
__global__ void dga_colstats(const u16b* __restrict__ qg, const u16b* __restrict__ kg,
                             float* __restrict__ colmax, float* __restrict__ colsum,
                             int E, int H, int mode) {
    const int wave = threadIdx.x >> 5, lane = threadIdx.x & 31;
    const int half = lane >> 4, r16 = lane & 15;
    const int sgroups = E >> 6;                        // 4 s-tiles per wave
    const int gid = blockIdx.x * 8 + wave;
    const int ti = gid / sgroups, sg = gid % sgroups;

    const u16b* aptr = qg + (long)(ti * 16 + r16) * H + half * 8;
    const u16b* bp0 = kg + (long)((sg * 4 + 0) * 16 + r16) * H + half * 8;
    const u16b* bp1 = kg + (long)((sg * 4 + 1) * 16 + r16) * H + half * 8;
    const u16b* bp2 = kg + (long)((sg * 4 + 2) * 16 + r16) * H + half * 8;
    const u16b* bp3 = kg + (long)((sg * 4 + 3) * 16 + r16) * H + half * 8;

    v8f a0 = {}, a1 = {}, a2 = {}, a3 = {};
#pragma unroll 4
    for (int k = 0; k < H; k += 32) {
        v16bf a = load_frag(aptr + k);
        __builtin_prefetch(aptr + k + 128, 0, 3);
        a0 = wmma_bf16(a, load_frag(bp0 + k), a0);
        a1 = wmma_bf16(a, load_frag(bp1 + k), a1);
        a2 = wmma_bf16(a, load_frag(bp2 + k), a2);
        a3 = wmma_bf16(a, load_frag(bp3 + k), a3);
    }
    v8f* accs[4] = {&a0, &a1, &a2, &a3};
#pragma unroll
    for (int j = 0; j < 4; ++j) {
        const int s = (sg * 4 + j) * 16 + r16;         // C layout: lane%16 = column
        v8f c = *accs[j];
        if (mode == 0) {
            float m = c[0];
#pragma unroll
            for (int r = 1; r < 8; ++r) m = fmaxf(m, c[r]);
            atomicMaxF(&colmax[s], m);
        } else {
            const float cm = colmax[s];
            float sum = 0.0f;
#pragma unroll
            for (int r = 0; r < 8; ++r) sum += __expf(c[r] - cm);
            atomicAdd(&colsum[s], sum);
        }
    }
}

// ---------- fused: recompute logits -> column softmax -> @ V -> scatter to tgt rows ----------
// Block = 8 waves owns 16 t-rows x all 512 h-cols.  Per 128-wide s-chunk:
//   phase 1: wave w computes its own 16x16 prob subtile (16 WMMAs), stores bf16 to LDS
//   phase 2: wave w updates its 64 h-cols: 4 ksteps x 4 n-tiles (16 WMMAs)
__global__ void dga_attn_out(const u16b* __restrict__ qg, const u16b* __restrict__ kg,
                             const u16b* __restrict__ vgT,
                             const float* __restrict__ colmax, const float* __restrict__ colsum,
                             const int* __restrict__ tgt, float* __restrict__ out,
                             int E, int H) {
    __shared__ __align__(16) u16b pbuf[16][128];       // 16 t-rows x 128 s-cols, bf16

    const int wave = threadIdx.x >> 5, lane = threadIdx.x & 31;
    const int half = lane >> 4, r16 = lane & 15;
    const int t0 = blockIdx.x * 16;
    const int nBase = wave * 64;                       // 8 waves * 64 = H = 512

    // q-tile A fragments resident in registers (16 ksteps over K=512)
    v16bf afr[16];
    {
        const u16b* aptr = qg + (long)(t0 + r16) * H + half * 8;
#pragma unroll
        for (int kk = 0; kk < 16; ++kk) afr[kk] = load_frag(aptr + kk * 32);
    }

    v8f acc[4] = {{}, {}, {}, {}};

    for (int s0 = 0; s0 < E; s0 += 128) {
        // ---- phase 1: probabilities for s in [s0 + wave*16, +16)
        const int sb = s0 + wave * 16;
        const u16b* bptr = kg + (long)(sb + r16) * H + half * 8;
        v8f lacc = {};
#pragma unroll
        for (int kk = 0; kk < 16; ++kk)
            lacc = wmma_bf16(afr[kk], load_frag(bptr + kk * 32), lacc);

        const int sAbs = sb + r16;                     // lane%16 = logits column
        const float cm = colmax[sAbs];
        const float inv = 1.0f / colsum[sAbs];
        const int scol = wave * 16 + r16;
#pragma unroll
        for (int r = 0; r < 8; ++r)
            pbuf[r + half * 8][scol] = f2bf(__expf(lacc[r] - cm) * inv);
        __syncthreads();

        // ---- phase 2: out[t, nBase..nBase+64) += P(16x128) @ vgT
#pragma unroll
        for (int kk = 0; kk < 4; ++kk) {
            union { uint4 q[2]; v16bf v; } af;         // A frag from LDS (ds_load_b128 x2)
            af.q[0] = *(const uint4*)&pbuf[r16][kk * 32 + half * 8];
            af.q[1] = *(const uint4*)&pbuf[r16][kk * 32 + 16 + half * 8];
#pragma unroll
            for (int nt = 0; nt < 4; ++nt) {
                const u16b* vb = vgT + (long)(nBase + nt * 16 + r16) * E
                                     + s0 + kk * 32 + half * 8;
                acc[nt] = wmma_bf16(af.v, load_frag(vb), acc[nt]);
            }
        }
        __builtin_prefetch(kg + (long)(s0 + 128 + wave * 16 + r16) * H, 0, 3);
        __syncthreads();
    }

    // scatter final rows into d_out at tgt[t]
#pragma unroll
    for (int nt = 0; nt < 4; ++nt) {
        const int col = nBase + nt * 16 + r16;
#pragma unroll
        for (int r = 0; r < 8; ++r) {
            const int t = t0 + r + half * 8;
            out[(long)tgt[t] * H + col] = acc[nt][r];
        }
    }
}

// ---------- host launcher ----------
extern "C" void kernel_launch(void* const* d_in, const int* in_sizes, int n_in,
                              void* d_out, int out_size, void* d_ws, size_t ws_size,
                              hipStream_t stream) {
    const float* x   = (const float*)d_in[0];
    const float* Wq  = (const float*)d_in[1];
    const float* bq  = (const float*)d_in[2];
    const float* Wk  = (const float*)d_in[3];
    const float* bk  = (const float*)d_in[4];
    const float* Wv  = (const float*)d_in[5];
    const float* bv  = (const float*)d_in[6];
    const int*  esrc = (const int*)d_in[7];
    const int*  etgt = (const int*)d_in[8];

    const int H = in_sizes[2];            // 512
    const int N = in_sizes[0] / H;        // 32768
    const int E = in_sizes[7];            // 8192
    float* out = (float*)d_out;

    // workspace carve-up (all naturally 256B aligned; ~58 MB total)
    char* w = (char*)d_ws;
    u16b*  x16  = (u16b*)w;  w += (size_t)N * H * 2;
    u16b*  wq16 = (u16b*)w;  w += (size_t)H * H * 2;
    u16b*  wk16 = (u16b*)w;  w += (size_t)H * H * 2;
    u16b*  wv16 = (u16b*)w;  w += (size_t)H * H * 2;
    u16b*  qg   = (u16b*)w;  w += (size_t)E * H * 2;   // bf16, pre-scaled by 1/sqrt(HEAD)
    u16b*  kg   = (u16b*)w;  w += (size_t)E * H * 2;
    u16b*  vgT  = (u16b*)w;  w += (size_t)H * E * 2;   // [H][E] K-major
    float* colmax = (float*)w; w += (size_t)E * 4;
    float* colsum = (float*)w; w += (size_t)E * 4;

    // 1) fp32 -> bf16 conversions
    dga_cvt_bf16<<<4096, 256, 0, stream>>>(x,  x16,  N * H);
    dga_cvt_bf16<<<256,  256, 0, stream>>>(Wq, wq16, H * H);
    dga_cvt_bf16<<<256,  256, 0, stream>>>(Wk, wk16, H * H);
    dga_cvt_bf16<<<256,  256, 0, stream>>>(Wv, wv16, H * H);
    dga_init_stats<<<(E + 255) / 256, 256, 0, stream>>>(colmax, colsum, E);

    // 2) projections (bf16 WMMA, f32 accumulate)
    const float qscale = 1.0f / sqrtf((float)(H / 8));  // 1/sqrt(HEAD) folded into q
    dga_proj<<<((N / 16) * (H / 16)) / 8, 256, 0, stream>>>(
        x16, wv16, bv, nullptr, H, 1.0f, out, nullptr);           // full V -> d_out
    dga_proj<<<((E / 16) * (H / 16)) / 8, 256, 0, stream>>>(
        x16, wq16, bq, etgt, H, qscale, nullptr, qg);             // q[tgt]
    dga_proj<<<((E / 16) * (H / 16)) / 8, 256, 0, stream>>>(
        x16, wk16, bk, esrc, H, 1.0f, nullptr, kg);               // k[src]

    // 3) gather + transpose V[src] to K-major bf16
    dga_gather_vT<<<(E * H) / 256, 256, 0, stream>>>(out, esrc, vgT, E, H);

    // 4) column softmax stats (logits recomputed, operands L2-resident)
    const int statBlocks = ((E / 16) * (E / 64)) / 8;
    dga_colstats<<<statBlocks, 256, 0, stream>>>(qg, kg, colmax, colsum, E, H, 0);
    dga_colstats<<<statBlocks, 256, 0, stream>>>(qg, kg, colmax, colsum, E, H, 1);

    // 5) fused logits -> softmax -> @V -> scatter into tgt rows of d_out
    dga_attn_out<<<E / 16, 256, 0, stream>>>(qg, kg, vgT, colmax, colsum, etgt, out, E, H);
}
